// CausalMultiheadSelfAttention_37563783970943
// MI455X (gfx1250) — compile-verified, matrix operand order checked by reading, not run on hardware
//
#include <hip/hip_runtime.h>

typedef __bf16 bf16_t;
typedef __bf16 v16bf __attribute__((ext_vector_type(16)));
typedef __bf16 v8bf  __attribute__((ext_vector_type(8)));
typedef __bf16 v4bf  __attribute__((ext_vector_type(4)));
typedef float  v8f   __attribute__((ext_vector_type(8)));

#define B_   2
#define S_   2048
#define D_   1024
#define H_   16
#define DK_  64
#define BS_  (B_ * S_)   // 4096 token rows

#define WMMA_BF16(a, b, c) \
  __builtin_amdgcn_wmma_f32_16x16x32_bf16(false, (a), false, (b), (short)0, (c), false, false)

static __device__ __forceinline__ v8f vzero8() {
  v8f v = {0.f, 0.f, 0.f, 0.f, 0.f, 0.f, 0.f, 0.f};
  return v;
}

// Load a 16x32 bf16 fragment (A layout; B layout is the transpose-symmetric
// case with identical per-lane addressing) from row-major memory.
// Per CDNA5 ISA: lane L(0-15) holds row L, K = {0..7, 16..23};
// lane L+16 holds row L, K = {8..15, 24..31}. Two 16-byte loads per lane.
static __device__ __forceinline__ v16bf load_frag(const bf16_t* base, int ld, int lane) {
  const bf16_t* p = base + (lane & 15) * ld + ((lane >> 4) << 3);
  v8bf lo = *(const v8bf*)p;
  v8bf hi = *(const v8bf*)(p + 16);
  union { v16bf v; v8bf h[2]; } u;
  u.h[0] = lo;
  u.h[1] = hi;
  return u.v;
}

// Reductions across the 16 lanes sharing one C-matrix row (xor masks < 16
// never cross the half-wave boundary).
static __device__ __forceinline__ float rmax16(float v) {
  v = fmaxf(v, __shfl_xor(v, 8, 32));
  v = fmaxf(v, __shfl_xor(v, 4, 32));
  v = fmaxf(v, __shfl_xor(v, 2, 32));
  v = fmaxf(v, __shfl_xor(v, 1, 32));
  return v;
}
static __device__ __forceinline__ float rsum16(float v) {
  v += __shfl_xor(v, 8, 32);
  v += __shfl_xor(v, 4, 32);
  v += __shfl_xor(v, 2, 32);
  v += __shfl_xor(v, 1, 32);
  return v;
}

// ---------------------------------------------------------------------------
// fp32 -> bf16 conversion, 4 elements per thread
// ---------------------------------------------------------------------------
__global__ void __launch_bounds__(256)
cvt4_kernel(const float* __restrict__ in, bf16_t* __restrict__ out, int n4) {
  int i = blockIdx.x * blockDim.x + threadIdx.x;
  if (i < n4) {
    float4 f = ((const float4*)in)[i];
    v4bf o;
    o.x = (bf16_t)f.x;
    o.y = (bf16_t)f.y;
    o.z = (bf16_t)f.z;
    o.w = (bf16_t)f.w;
    ((v4bf*)out)[i] = o;
  }
}

// ---------------------------------------------------------------------------
// Fully-unrolled 16x64-per-wave GEMM body: C(16x64) += A(16x1024) B^T.
// No backedge -> no register rotation copies; loads land directly in the
// VGPR octets their consuming WMMA reads, scheduler hoists across K-steps.
// ---------------------------------------------------------------------------
static __device__ __forceinline__ void gemm_rowtile(const bf16_t* __restrict__ arow,
                                                    const bf16_t* __restrict__ wrow,
                                                    int lane, v8f acc[4]) {
#pragma unroll
  for (int kk = 0; kk < D_ / 32; ++kk) {
    const int k0 = kk * 32;
    v16bf a = load_frag(arow + k0, D_, lane);
    v16bf b0 = load_frag(wrow + 0 * 16 * D_ + k0, D_, lane);
    v16bf b1 = load_frag(wrow + 1 * 16 * D_ + k0, D_, lane);
    v16bf b2 = load_frag(wrow + 2 * 16 * D_ + k0, D_, lane);
    v16bf b3 = load_frag(wrow + 3 * 16 * D_ + k0, D_, lane);
    acc[0] = WMMA_BF16(a, b0, acc[0]);
    acc[1] = WMMA_BF16(a, b1, acc[1]);
    acc[2] = WMMA_BF16(a, b2, acc[2]);
    acc[3] = WMMA_BF16(a, b3, acc[3]);
  }
}

// ---------------------------------------------------------------------------
// Fused QKV projection: y = x @ W^T for W in {Wq, Wk, Wv}.
// Q, K written [B,H,S,dk]; V written transposed [B,H,dk,S].
// grid = (64, 16 heads, 3 matrices), block = 128 (4 waves)
// ---------------------------------------------------------------------------
__global__ void __launch_bounds__(128)
proj_qkv_kernel(const bf16_t* __restrict__ xb,
                const bf16_t* __restrict__ Wqb,
                const bf16_t* __restrict__ Wkb,
                const bf16_t* __restrict__ Wvb,
                bf16_t* __restrict__ Qb,
                bf16_t* __restrict__ Kb,
                bf16_t* __restrict__ Vtb) {
  const int lane  = threadIdx.x & 31;
  const int wave  = threadIdx.x >> 5;
  const int mtile = blockIdx.x * 4 + wave;   // 0..255
  const int row0  = mtile * 16;              // token row
  const int h     = blockIdx.y;              // 64-wide N tile == head
  const int which = blockIdx.z;              // 0=Q 1=K 2=V

  const bf16_t* W    = (which == 0) ? Wqb : ((which == 1) ? Wkb : Wvb);
  const bf16_t* arow = xb + (size_t)row0 * D_;
  const bf16_t* wrow = W + (size_t)(h * DK_) * D_;

  v8f acc[4];
#pragma unroll
  for (int t = 0; t < 4; ++t) acc[t] = vzero8();

  gemm_rowtile(arow, wrow, lane, acc);

  const int roff = (lane >> 4) << 3;  // 0 or 8
  const int col  = lane & 15;
#pragma unroll
  for (int r = 0; r < 8; ++r) {
    const int rowg = row0 + roff + r;   // 0..4095
    const int bb   = rowg >> 11;        // / 2048
    const int s    = rowg & (S_ - 1);
#pragma unroll
    for (int t = 0; t < 4; ++t) {
      const bf16_t v = (bf16_t)acc[t][r];
      const int d = t * 16 + col;
      if (which == 0)
        Qb[((bb * H_ + h) * S_ + s) * DK_ + d] = v;
      else if (which == 1)
        Kb[((bb * H_ + h) * S_ + s) * DK_ + d] = v;
      else
        Vtb[((bb * H_ + h) * DK_ + d) * S_ + s] = v;
    }
  }
}

// ---------------------------------------------------------------------------
// Flash-style causal attention. One wave owns a 16-row query tile; iterates
// 32-key blocks up to the diagonal with an online softmax. All K and V
// fragment loads are issued at the top of each iteration so V is in flight
// during the score WMMAs and the softmax VALU work. P tile is staged through
// LDS to convert C-matrix layout -> A-fragment layout for P·V.
// grid = (16, 16 heads, 2 batch), block = 256 (8 waves), 8 KB LDS
// ---------------------------------------------------------------------------
__global__ void __launch_bounds__(256)
attn_kernel(const bf16_t* __restrict__ Qb,
            const bf16_t* __restrict__ Kb,
            const bf16_t* __restrict__ Vtb,
            bf16_t* __restrict__ Ob) {
  __shared__ bf16_t plds[8][16 * 32];

  const int lane  = threadIdx.x & 31;
  const int wave  = threadIdx.x >> 5;
  const int h     = blockIdx.y;
  const int b     = blockIdx.z;
  const int qtile = blockIdx.x * 8 + wave;   // 0..127
  const int qbase = qtile * 16;

  const bf16_t* Qp  = Qb  + ((size_t)(b * H_ + h) * S_) * DK_;
  const bf16_t* Kp  = Kb  + ((size_t)(b * H_ + h) * S_) * DK_;
  const bf16_t* Vtp = Vtb + ((size_t)(b * H_ + h) * DK_) * S_;

  const v16bf aq0 = load_frag(Qp + qbase * DK_ + 0, DK_, lane);
  const v16bf aq1 = load_frag(Qp + qbase * DK_ + 32, DK_, lane);

  const int roff = (lane >> 4) << 3;  // 0 or 8
  const int col  = lane & 15;

  float m_i[8], l_i[8];
  v8f acc[4];
#pragma unroll
  for (int r = 0; r < 8; ++r) { m_i[r] = -__builtin_inff(); l_i[r] = 0.f; }
#pragma unroll
  for (int t = 0; t < 4; ++t) acc[t] = vzero8();

  // softmax in exp2 domain: 1/sqrt(dk) * log2(e)
  const float sm_scale = 0.125f * 1.44269504088896f;

  const int nblocks = (qbase + 47) >> 5;
  for (int jb = 0; jb < nblocks; ++jb) {
    const int jbase = jb * 32;

    // --- issue ALL fragment loads for this block up front
    v16bf bk00 = load_frag(Kp + (jbase + 0) * DK_ + 0, DK_, lane);
    v16bf bk01 = load_frag(Kp + (jbase + 0) * DK_ + 32, DK_, lane);
    v16bf bk10 = load_frag(Kp + (jbase + 16) * DK_ + 0, DK_, lane);
    v16bf bk11 = load_frag(Kp + (jbase + 16) * DK_ + 32, DK_, lane);
    v16bf bv[4];
#pragma unroll
    for (int t = 0; t < 4; ++t) bv[t] = load_frag(Vtp + (t * 16) * S_ + jbase, S_, lane);

    // --- scores: two 16-wide key subtiles, K-dim = dk = 64 (2 WMMAs each)
    v8f c0 = vzero8();
    v8f c1 = vzero8();
    c0 = WMMA_BF16(aq0, bk00, c0);
    c0 = WMMA_BF16(aq1, bk01, c0);
    c1 = WMMA_BF16(aq0, bk10, c1);
    c1 = WMMA_BF16(aq1, bk11, c1);

    // --- causal mask + online softmax update, stage P tile into LDS
#pragma unroll
    for (int r = 0; r < 8; ++r) {
      const int rowg = qbase + roff + r;
      float s0 = ((jbase + col) <= rowg) ? c0[r] * sm_scale : -__builtin_inff();
      float s1 = ((jbase + 16 + col) <= rowg) ? c1[r] * sm_scale : -__builtin_inff();
      float mx = rmax16(fmaxf(s0, s1));
      float mnew = fmaxf(m_i[r], mx);
      float corr = exp2f(m_i[r] - mnew);
      float e0 = exp2f(s0 - mnew);
      float e1 = exp2f(s1 - mnew);
      float rsum = rsum16(e0 + e1);
      l_i[r] = l_i[r] * corr + rsum;
      m_i[r] = mnew;
#pragma unroll
      for (int t = 0; t < 4; ++t) acc[t][r] *= corr;
      plds[wave][(roff + r) * 32 + col]      = (bf16_t)e0;
      plds[wave][(roff + r) * 32 + 16 + col] = (bf16_t)e1;
    }

    // per-wave LDS RAW fence (no block barrier: waves have divergent trips)
    __asm volatile("s_wait_dscnt 0" ::: "memory");

    // --- P(16x32) @ V(32x64): A-frag from LDS, B-frags preloaded above
    v16bf ap = load_frag(&plds[wave][0], 32, lane);
#pragma unroll
    for (int t = 0; t < 4; ++t) acc[t] = WMMA_BF16(ap, bv[t], acc[t]);
  }

  // --- normalize and emit [B*S, D] bf16 (heads re-interleaved)
#pragma unroll
  for (int r = 0; r < 8; ++r) {
    const float inv = 1.0f / l_i[r];
    const int rowg = b * S_ + qbase + roff + r;
#pragma unroll
    for (int t = 0; t < 4; ++t) {
      Ob[(size_t)rowg * D_ + h * DK_ + t * 16 + col] = (bf16_t)(acc[t][r] * inv);
    }
  }
}

// ---------------------------------------------------------------------------
// Output projection: out = O @ Wo^T, fp32 result straight to d_out.
// grid = (64, 16), block = 128 (4 waves), one wave per 16x64 tile
// ---------------------------------------------------------------------------
__global__ void __launch_bounds__(128)
out_proj_kernel(const bf16_t* __restrict__ Ob,
                const bf16_t* __restrict__ Wob,
                float* __restrict__ out) {
  const int lane  = threadIdx.x & 31;
  const int wave  = threadIdx.x >> 5;
  const int mtile = blockIdx.x * 4 + wave;   // 0..255
  const int row0  = mtile * 16;
  const int n64   = blockIdx.y;

  const bf16_t* arow = Ob + (size_t)row0 * D_;
  const bf16_t* wrow = Wob + (size_t)(n64 * 64) * D_;

  v8f acc[4];
#pragma unroll
  for (int t = 0; t < 4; ++t) acc[t] = vzero8();

  gemm_rowtile(arow, wrow, lane, acc);

  const int roff = (lane >> 4) << 3;
  const int col  = lane & 15;
#pragma unroll
  for (int r = 0; r < 8; ++r) {
    const int rowg = row0 + roff + r;
#pragma unroll
    for (int t = 0; t < 4; ++t) {
      out[(size_t)rowg * D_ + n64 * 64 + t * 16 + col] = acc[t][r];
    }
  }
}

// ---------------------------------------------------------------------------
extern "C" void kernel_launch(void* const* d_in, const int* in_sizes, int n_in,
                              void* d_out, int out_size, void* d_ws, size_t ws_size,
                              hipStream_t stream) {
  const float* x  = (const float*)d_in[0];
  const float* Wq = (const float*)d_in[1];
  const float* Wk = (const float*)d_in[2];
  const float* Wv = (const float*)d_in[3];
  const float* Wo = (const float*)d_in[4];
  float* out = (float*)d_out;

  char* ws = (char*)d_ws;
  bf16_t* xb  = (bf16_t*)(ws + 0);                       //  8 MB: x bf16
  bf16_t* Wqb = (bf16_t*)(ws + (8u << 20));              //  2 MB each
  bf16_t* Wkb = (bf16_t*)(ws + (10u << 20));
  bf16_t* Wvb = (bf16_t*)(ws + (12u << 20));
  bf16_t* Wob = (bf16_t*)(ws + (14u << 20));
  bf16_t* Qb  = (bf16_t*)(ws + (16u << 20));             //  8 MB [B,H,S,dk]
  bf16_t* Kb  = (bf16_t*)(ws + (24u << 20));             //  8 MB [B,H,S,dk]
  bf16_t* Vtb = (bf16_t*)(ws + (32u << 20));             //  8 MB [B,H,dk,S]
  bf16_t* Ob  = (bf16_t*)(ws + (40u << 20));             //  8 MB [B*S,D]

  cvt4_kernel<<<(BS_ * D_ / 4 + 255) / 256, 256, 0, stream>>>(x, xb, BS_ * D_ / 4);
  cvt4_kernel<<<(D_ * D_ / 4 + 255) / 256, 256, 0, stream>>>(Wq, Wqb, D_ * D_ / 4);
  cvt4_kernel<<<(D_ * D_ / 4 + 255) / 256, 256, 0, stream>>>(Wk, Wkb, D_ * D_ / 4);
  cvt4_kernel<<<(D_ * D_ / 4 + 255) / 256, 256, 0, stream>>>(Wv, Wvb, D_ * D_ / 4);
  cvt4_kernel<<<(D_ * D_ / 4 + 255) / 256, 256, 0, stream>>>(Wo, Wob, D_ * D_ / 4);

  proj_qkv_kernel<<<dim3(64, 16, 3), 128, 0, stream>>>(xb, Wqb, Wkb, Wvb, Qb, Kb, Vtb);
  attn_kernel<<<dim3(16, 16, 2), 256, 0, stream>>>(Qb, Kb, Vtb, Ob);
  out_proj_kernel<<<dim3(64, 16, 1), 128, 0, stream>>>(Ob, Wob, out);
}
